// PacMan_69441031242525
// MI455X (gfx1250) — matrix operand order repaired and checked by reading, not compiled, not run on hardware
//
#include <hip/hip_runtime.h>

typedef __attribute__((ext_vector_type(16))) _Float16 v16h;
typedef __attribute__((ext_vector_type(8)))  _Float16 v8h;
typedef __attribute__((ext_vector_type(8)))  float    v8f;

#define K_KICK   10
#define NE       60
#define TOF      16
#define NPH      80
#define NKE      60
#define NPIX     960           // NE*TOF (row length)
#define NATOM    4800          // NPH*NKE (rows)
#define NTILE    300           // NATOM/16
#define COST_ITS 100
#define EAT_FAC  0.05f
#define BATCH    2
#define BLOCK    1024
#define NWAVE    (BLOCK/32)

// ---------------------------------------------------------------------------
// basis[k][i][p][e][d] (f32)  ->  redbh[k][p*NKE+i][e*TOF+d] (f16, row-major)
// One HBM pass; afterwards all 2200 GEMV steps stream the 92MB f16 copy,
// which fits entirely in the 192MB L2.
// ---------------------------------------------------------------------------
__global__ void cvt_kernel(const float* __restrict__ basis,
                           _Float16* __restrict__ redbh, long n) {
  long idx = (long)blockIdx.x * blockDim.x + threadIdx.x;
  if (idx >= n) return;
  int d = (int)(idx % TOF);      long t = idx / TOF;
  int e = (int)(t % NE);         t /= NE;
  int p = (int)(t % NPH);        t /= NPH;
  int i = (int)(t % NKE);        t /= NKE;
  int kk = (int)t;
  long out = ((long)kk * NATOM + (long)p * NKE + i) * NPIX + e * TOF + d;
  redbh[out] = (_Float16)basis[idx];
}

// ---------------------------------------------------------------------------
// One matching-pursuit step: scores = rows(f16) @ acgh(f16) via WMMA f16.
// A layout (16-bit 16x32): lane L (m=L&15, g=L>>4) holds row (rbase+m),
//   halves [0..15] = K = g*16 + 0..15 (contiguous, matching B).
// B layout (16-bit 32x16): lane L holds 16 contiguous K at g*16; all 16
//   N-columns replicate acg, so the value depends only on K.
// D: lane L, VGPR r = D[M = r + 8*g][N = L&15]; columns identical -> every
//   lane holds 8 of the tile's 16 scores in registers.
// Argmax: per-lane running max is branchless (ascending idx => strict '>'
// keeps the first max). Scores are >= 0, so (score_bits<<32)|(NATOM-idx) is a
// sortable u64 whose max realizes jnp.argmax first-occurrence semantics.
// ---------------------------------------------------------------------------
__device__ __forceinline__ int matvec_argmax(const _Float16* __restrict__ rowsh,
                                             const _Float16* acgh,
                                             unsigned long long* redp,
                                             int* bestShared, int tid) {
  const int lane = tid & 31;
  const int wave = tid >> 5;
  const int g    = lane >> 4;
  const int m    = lane & 15;

  float bs = -1.0f;    // scores >= 0 and every lane visits >= 72 scores,
  int   bi = 0;        // so the sentinel is always overwritten before packing

  for (int tile = wave; tile < NTILE; tile += NWAVE) {
    const int rbase = tile * 16;
    const _Float16* arow = rowsh + (size_t)(rbase + m) * NPIX;
    __builtin_prefetch(arow, 0, 0);            // global_prefetch_b8
    __builtin_prefetch(arow + 480, 0, 0);
    v8f c = {};
#pragma unroll 6
    for (int kc = 0; kc < NPIX; kc += 32) {
      v16h a = *(const v16h*)(arow + kc + g * 16);
      v16h b = *(const v16h*)(acgh + kc + g * 16);
      c = __builtin_amdgcn_wmma_f32_16x16x32_f16(false, a, false, b,
                                                 (short)0, c, false, false);
    }
#pragma unroll
    for (int r = 0; r < 8; ++r) {              // branchless: cndmask + max
      float s   = c[r];
      int   idx = rbase + 8 * g + r;
      bi = (s > bs) ? idx : bi;
      bs = fmaxf(bs, s);
    }
  }
  // pack into sortable u64: max => (max score, then min index)
  unsigned long long pk =
      ((unsigned long long)__float_as_uint(bs) << 32) |
      (unsigned int)(NATOM - bi);
#pragma unroll
  for (int off = 16; off > 0; off >>= 1) {
    unsigned long long o = __shfl_down(pk, off, 32);
    pk = (o > pk) ? o : pk;
  }
  if (lane == 0) redp[wave] = pk;
  __syncthreads();
  if (wave == 0) {
    unsigned long long p2 = redp[lane];
#pragma unroll
    for (int off = 16; off > 0; off >>= 1) {
      unsigned long long o = __shfl_down(p2, off, 32);
      p2 = (o > p2) ? o : p2;
    }
    if (lane == 0) *bestShared = NATOM - (int)(unsigned int)(p2 & 0xffffffffull);
  }
  __syncthreads();
  return *bestShared;
}

// ---------------------------------------------------------------------------
// Cost phase: one workgroup per (batch, kick) chain. 100 steps, full subtract,
// then residual energy -> costs[b*K + k].
// ---------------------------------------------------------------------------
__global__ __launch_bounds__(BLOCK) void cost_kernel(
    const float* __restrict__ img, const float* __restrict__ basis,
    const _Float16* __restrict__ redbh, float* __restrict__ costs) {
  __shared__ float acg[NPIX];
  __shared__ __align__(64) _Float16 acgh[NPIX];
  __shared__ unsigned long long redp[NWAVE];
  __shared__ float redv[NWAVE];
  __shared__ int   bestShared;

  const int tid = threadIdx.x;
  const int b   = blockIdx.x / K_KICK;
  const int kk  = blockIdx.x % K_KICK;
  const _Float16* rowsh = redbh + (size_t)kk * NATOM * NPIX;

  for (int j = tid; j < NPIX; j += BLOCK) {
    float v = img[b * NPIX + j];
    acg[j] = v; acgh[j] = (_Float16)v;
  }
  __syncthreads();

  for (int it = 0; it < COST_ITS; ++it) {
    int best = matvec_argmax(rowsh, acgh, redp, &bestShared, tid);
    int p = best / NKE, i = best % NKE;        // row index -> basis[k][i][p][:][:]
    const float* row = basis + ((size_t)(kk * NKE + i) * NPH + p) * NPIX;
    for (int j = tid; j < NPIX; j += BLOCK) {  // fused update + f16 convert
      float v = fmaxf(acg[j] - row[j], 0.0f);
      acg[j] = v; acgh[j] = (_Float16)v;
    }
    __syncthreads();
  }

  float ps = 0.0f;
  for (int j = tid; j < NPIX; j += BLOCK) ps += acg[j] * acg[j];
#pragma unroll
  for (int off = 16; off > 0; off >>= 1) ps += __shfl_down(ps, off, 32);
  if ((tid & 31) == 0) redv[tid >> 5] = ps;
  __syncthreads();
  if (tid == 0) {
    float s = 0.0f;
    for (int w = 0; w < NWAVE; ++w) s += redv[w];
    costs[b * K_KICK + kk] = s;
  }
}

__global__ void argmin_kernel(const float* __restrict__ costs,
                              int* __restrict__ sel) {
  int b = threadIdx.x;
  if (b < BATCH) {
    float bv = costs[b * K_KICK]; int bi = 0;
    for (int k = 1; k < K_KICK; ++k) {
      float v = costs[b * K_KICK + k];
      if (v < bv) { bv = v; bi = k; }          // first-min like jnp.argmin
    }
    sel[b] = bi;
  }
}

// ---------------------------------------------------------------------------
// Rec phase: one workgroup per batch sample. its steps with EAT_FAC subtract;
// accumulate image, histogram counts, final residual. Outputs concatenated:
// [images (B,60,16)] [histo.T (B,60,80)] [acg (B,60,16)].
// ---------------------------------------------------------------------------
__global__ __launch_bounds__(BLOCK) void rec_kernel(
    const float* __restrict__ img, const float* __restrict__ basis,
    const _Float16* __restrict__ redbh, const int* __restrict__ sel,
    const int* __restrict__ its_ptr, float* __restrict__ out) {
  __shared__ float acg[NPIX];
  __shared__ __align__(64) _Float16 acgh[NPIX];
  __shared__ unsigned long long redp[NWAVE];
  __shared__ int   bestShared;
  __shared__ float image[NPIX];
  __shared__ int   histo[NATOM];

  const int tid = threadIdx.x;
  const int b   = blockIdx.x;
  const int kk  = sel[b];
  const int its = its_ptr[0];
  const _Float16* rowsh = redbh + (size_t)kk * NATOM * NPIX;

  for (int j = tid; j < NPIX; j += BLOCK) {
    float v = img[b * NPIX + j];
    acg[j] = v; acgh[j] = (_Float16)v; image[j] = 0.0f;
  }
  for (int j = tid; j < NATOM; j += BLOCK) histo[j] = 0;
  __syncthreads();

  for (int it = 0; it < its; ++it) {
    int best = matvec_argmax(rowsh, acgh, redp, &bestShared, tid);
    int p = best / NKE, i = best % NKE;
    const float* row = basis + ((size_t)(kk * NKE + i) * NPH + p) * NPIX;
    if (tid == 0) histo[best] += 1;
    for (int j = tid; j < NPIX; j += BLOCK) {
      float rv = row[j];
      image[j] += rv;
      float v = fmaxf(acg[j] - EAT_FAC * rv, 0.0f);
      acg[j] = v; acgh[j] = (_Float16)v;
    }
    __syncthreads();
  }

  float* out_img  = out + (size_t)b * NPIX;
  float* out_hist = out + (size_t)BATCH * NPIX + (size_t)b * NATOM;
  float* out_acg  = out + (size_t)BATCH * NPIX + (size_t)BATCH * NATOM + (size_t)b * NPIX;
  for (int j = tid; j < NPIX; j += BLOCK) { out_img[j] = image[j]; out_acg[j] = acg[j]; }
  // histo.T: element (i,p) of (NKE,NPH) = counts[p*NKE + i]
  for (int j = tid; j < NATOM; j += BLOCK) {
    int i2 = j / NPH, p2 = j % NPH;
    out_hist[j] = (float)histo[p2 * NKE + i2];
  }
}

// ---------------------------------------------------------------------------
extern "C" void kernel_launch(void* const* d_in, const int* in_sizes, int n_in,
                              void* d_out, int out_size, void* d_ws, size_t ws_size,
                              hipStream_t stream) {
  (void)in_sizes; (void)n_in; (void)out_size; (void)ws_size;
  const float* img   = (const float*)d_in[0];
  const float* basis = (const float*)d_in[1];
  const int*   its   = (const int*)d_in[2];
  float*       out   = (float*)d_out;

  // workspace layout: [costs: 20 f32][sel: 2 i32 @ +256][f16 basis @ +512]
  float*     costs = (float*)d_ws;
  int*       sel   = (int*)((char*)d_ws + 256);
  _Float16*  redbh = (_Float16*)((char*)d_ws + 512);

  const long n = (long)K_KICK * NKE * NPH * NPIX;   // 46,080,000 elements
  cvt_kernel<<<(int)((n + 511) / 512), 512, 0, stream>>>(basis, redbh, n);
  cost_kernel<<<BATCH * K_KICK, BLOCK, 0, stream>>>(img, basis, redbh, costs);
  argmin_kernel<<<1, 64, 0, stream>>>(costs, sel);
  rec_kernel<<<BATCH, BLOCK, 0, stream>>>(img, basis, redbh, sel, its, out);
}
